// DenoisedUpsampling_77841987273302
// MI455X (gfx1250) — matrix operand order repaired and checked by reading, not compiled, not run on hardware
//
#include <hip/hip_runtime.h>
#include <hip/hip_bf16.h>

typedef _Float16 f16;
typedef _Float16 half16 __attribute__((ext_vector_type(16)));
typedef float    float8 __attribute__((ext_vector_type(8)));

union H16 { half16 v; uint4 q[2]; f16 e[16]; };
union F8  { float8 v; float e[8]; };

#define BN_EPS 1e-5f
#define LEAK 0.05f

__device__ inline float8 wmma_f16(half16 a, half16 b, float8 c) {
  // D = A(16x32 f16) * B(32x16 f16) + C(16x16 f32)
  return __builtin_amdgcn_wmma_f32_16x16x32_f16(false, a, false, b, (short)0, c, false, false);
}

// Load A fragment: lane holds row (lane&15); halves e[0..7] <- k0+hi*8.., e[8..15] <- k0+16+hi*8..
__device__ inline half16 load_a_frag(const f16* __restrict__ Arow, int k0, int hi) {
  H16 a;
  a.q[0] = *(const uint4*)(Arow + k0 + hi * 8);
  a.q[1] = *(const uint4*)(Arow + k0 + 16 + hi * 8);
  return a.v;
}

// Load B fragment from pre-packed weights (16 contiguous halves per lane)
__device__ inline half16 load_b_frag(const f16* __restrict__ Wl, int kb) {
  H16 b;
  const uint4* bp = (const uint4*)(Wl + (size_t)kb * 512);
  b.q[0] = bp[0];
  b.q[1] = bp[1];
  return b.v;
}

// -------------------------------------------------- weight pre-pack (one-time)
// Wp[((tn*nK0 + kb)*32 + lane)*16 + j] = W[(kb*32 + (j>>3)*16 + (lane>>4)*8 + (j&7))*N + (lane&15)+tn*16]
// zero-padded for k >= K. Makes every B-fragment two b128 loads.
__global__ void pack_b_kernel(const float* __restrict__ W, f16* __restrict__ Wp,
                              int K, int N, int Kp) {
  size_t i = (size_t)blockIdx.x * blockDim.x + threadIdx.x;
  int j = (int)(i & 15);
  int lane = (int)((i >> 4) & 31);
  size_t rest = i >> 9;
  int nK0 = Kp >> 5;
  int kb = (int)(rest % nK0);
  int tn = (int)(rest / nK0);
  int k = (kb << 5) + ((j >> 3) << 4) + ((lane >> 4) << 3) + (j & 7);
  int col = (lane & 15) + (tn << 4);
  Wp[i] = (k < K) ? (f16)W[(size_t)k * N + col] : (f16)0;
}

// ---------------------------------------------------------------- KNN (top-16)
__global__ __launch_bounds__(256)
void knn_kernel(const float* __restrict__ x, int* __restrict__ idx, int N) {
  __shared__ float xs[4096 * 3];
  int b = blockIdx.x >> 4;          // N/256 == 16 query blocks per batch
  int qblk = blockIdx.x & 15;
  const float* xb = x + (size_t)b * N * 3;
  for (int i = threadIdx.x; i < N * 3; i += 256) xs[i] = xb[i];
  __syncthreads();
  int q = qblk * 256 + threadIdx.x;
  float qx = xs[q * 3 + 0], qy = xs[q * 3 + 1], qz = xs[q * 3 + 2];
  float bd[16]; int bi[16];
#pragma unroll
  for (int r = 0; r < 16; ++r) { bd[r] = 3.4e38f; bi[r] = 0x7fffffff; }
  for (int j = 0; j < N; ++j) {
    float dx = qx - xs[j * 3 + 0];
    float dy = qy - xs[j * 3 + 1];
    float dz = qz - xs[j * 3 + 2];
    float d = dx * dx + dy * dy + dz * dz;
    if (d < bd[15] || (d == bd[15] && j < bi[15])) {
      int p = 15;
      while (p > 0 && (d < bd[p - 1] || (d == bd[p - 1] && j < bi[p - 1]))) {
        bd[p] = bd[p - 1]; bi[p] = bi[p - 1]; --p;
      }
      bd[p] = d; bi[p] = j;
    }
  }
  int* op = idx + ((size_t)b * N + q) * 16;
#pragma unroll
  for (int r = 0; r < 16; ++r) op[r] = bi[r];
}

// ------------------------------------------------ fused edge-conv encoder (WMMA)
// One block = one point; 16 neighbors = M dimension. DenseNet feature matrix in
// LDS (16 rows x 160 f16, zero-initialized so k-padding contributes 0).
__global__ __launch_bounds__(64)
void encoder_kernel(const float* __restrict__ x, const int* __restrict__ idx,
                    const f16* __restrict__ wp0, const f16* __restrict__ wp1,
                    const f16* __restrict__ wp2, const f16* __restrict__ wp3,
                    const float* __restrict__ bnscale, const float* __restrict__ bnshift,
                    const f16* __restrict__ wpout, const float* __restrict__ bout,
                    f16* __restrict__ out /* (B*N, 256) cols 0..127 */, int N) {
  __shared__ f16 F[16 * 160];
  int p = blockIdx.x;
  int b = p / N, n = p % N;
  int t = threadIdx.x;
  for (int i = t; i < 16 * 160; i += 64) F[i] = (f16)0;   // zero padding cols
  __syncthreads();
  if (t < 16) {
    int nb = idx[(size_t)p * 16 + t];
    const float* xq = x + ((size_t)b * N + n) * 3;
    const float* xk = x + ((size_t)b * N + nb) * 3;
    float ax = xq[0], ay = xq[1], az = xq[2];
    float kx = xk[0], ky = xk[1], kz = xk[2];
    f16* row = F + t * 160;
    row[0] = (f16)ax; row[1] = (f16)ay; row[2] = (f16)az;
    row[3] = (f16)kx; row[4] = (f16)ky; row[5] = (f16)kz;
    row[6] = (f16)(kx - ax); row[7] = (f16)(ky - ay); row[8] = (f16)(kz - az);
  }
  __syncthreads();

  int w = t >> 5;         // wave 0..1 (wave32)
  int lane = t & 31;
  int hi = lane >> 4;
  const f16* Arow = F + (lane & 15) * 160;
  int col = (lane & 15) + 16 * w;

  const f16* Wps[4] = {wp0, wp1, wp2, wp3};
  const int nK0s[4] = {1, 2, 3, 4};       // ceil32(9,41,73,105)

  for (int ci = 0; ci < 4; ++ci) {
    int nK0 = nK0s[ci];
    const f16* Wl = Wps[ci] + (((size_t)w * nK0) * 32 + lane) * 16;
    F8 acc;
#pragma unroll
    for (int r = 0; r < 8; ++r) acc.e[r] = 0.f;
    for (int kb = 0; kb < nK0; ++kb)
      acc.v = wmma_f16(load_a_frag(Arow, kb << 5, hi), load_b_frag(Wl, kb), acc.v);
    float sc = bnscale[ci * 32 + col];
    float sh = bnshift[ci * 32 + col];
#pragma unroll
    for (int r = 0; r < 8; ++r) {
      int rowM = r + hi * 8;
      float h = acc.e[r] * sc + sh;
      h = (h > 0.f) ? h : LEAK * h;
      F[rowM * 160 + 9 + ci * 32 + col] = (f16)h;
    }
    __syncthreads();
  }

  // 137 -> 128 projection (Kp = 160), then max over the 16 neighbor rows
  const int nK0 = 5;
  F8 acc4[4];
#pragma unroll
  for (int tt = 0; tt < 4; ++tt)
#pragma unroll
    for (int r = 0; r < 8; ++r) acc4[tt].e[r] = 0.f;
  for (int kb = 0; kb < nK0; ++kb) {
    half16 a = load_a_frag(Arow, kb << 5, hi);
#pragma unroll
    for (int tt = 0; tt < 4; ++tt) {
      int tn = w * 4 + tt;
      const f16* Wl = wpout + (((size_t)tn * nK0) * 32 + lane) * 16;
      acc4[tt].v = wmma_f16(a, load_b_frag(Wl, kb), acc4[tt].v);
    }
  }
#pragma unroll
  for (int tt = 0; tt < 4; ++tt) {
    int colg = (lane & 15) + 16 * (w * 4 + tt);
    float m = -3.4e38f;
#pragma unroll
    for (int r = 0; r < 8; ++r) m = fmaxf(m, acc4[tt].e[r]);
    m += bout[colg];
    float o = __shfl_xor(m, 16, 32);     // rows 8..15 live in lanes 16..31
    m = fmaxf(m, o);
    if (lane < 16) out[(size_t)p * 256 + colg] = (f16)m;
  }
}

// ---------------------------------------------------- generic WMMA GEMM kernel
// C[M,N] = act(A[M,Kp] (f16, lda >= Kp, 32-mult) x Wp(packed, zero-padded) + bias)
// one 16x16 tile per wave; inner loop: 4x b128 loads + 1 wmma, branch-free.
__global__ __launch_bounds__(128)
void gemm16(const f16* __restrict__ A, int lda, const f16* __restrict__ Wp,
            const float* __restrict__ bias, float* __restrict__ Cf,
            f16* __restrict__ Ch, int ldc, int M, int Kp, int N, int relu) {
  int wid = threadIdx.x >> 5;
  int lane = threadIdx.x & 31;
  int tilesN = N >> 4;
  int tile = blockIdx.x * 4 + wid;
  if (tile >= (M >> 4) * tilesN) return;   // wave-uniform -> EXEC stays full
  int tm = tile / tilesN, tn = tile % tilesN;
  int hi = lane >> 4;
  int rowA = (tm << 4) + (lane & 15);
  int col = (tn << 4) + (lane & 15);
  int nK0 = Kp >> 5;
  const f16* Arow = A + (size_t)rowA * lda;
  const f16* Wl = Wp + (((size_t)tn * nK0) * 32 + lane) * 16;
  F8 acc;
#pragma unroll
  for (int r = 0; r < 8; ++r) acc.e[r] = 0.f;
  for (int kb = 0; kb < nK0; ++kb)
    acc.v = wmma_f16(load_a_frag(Arow, kb << 5, hi), load_b_frag(Wl, kb), acc.v);
  float bcol = bias ? bias[col] : 0.f;
#pragma unroll
  for (int r = 0; r < 8; ++r) {
    int row = (tm << 4) + r + hi * 8;
    float hv = acc.e[r] + bcol;
    if (relu) hv = fmaxf(hv, 0.f);
    if (Cf) Cf[(size_t)row * ldc + col] = hv;
    if (Ch) Ch[(size_t)row * ldc + col] = (f16)hv;
  }
}

// ------------------------------------------------------------ small helpers
__global__ void bn_fold_kernel(const float* __restrict__ bc, const float* __restrict__ g,
                               const float* __restrict__ be, const float* __restrict__ m,
                               const float* __restrict__ v, float* __restrict__ scale,
                               float* __restrict__ shift) {
  int c = threadIdx.x;
  if (c < 32) {
    float s = g[c] * rsqrtf(v[c] + BN_EPS);
    scale[c] = s;
    shift[c] = (bc[c] - m[c]) * s + be[c];
  }
}

__global__ __launch_bounds__(256)
void mlp1_kernel(const float* __restrict__ gf, const float* __restrict__ w1,
                 const float* __restrict__ b1, const float* __restrict__ w2,
                 const float* __restrict__ b2, float* __restrict__ gout) {
  __shared__ float gin[512];
  __shared__ float h[256];
  int b = blockIdx.x, t = threadIdx.x;
  for (int i = t; i < 512; i += 256) gin[i] = gf[b * 512 + i];
  __syncthreads();
  float acc = b1[t];
  for (int k = 0; k < 512; ++k) acc += gin[k] * w1[k * 256 + t];
  h[t] = 0.5f * acc * (1.f + erff(acc * 0.7071067811865475f));   // exact GELU
  __syncthreads();
  if (t < 128) {
    float a2 = b2[t];
    for (int k = 0; k < 256; ++k) a2 += h[k] * w2[k * 128 + t];
    gout[b * 128 + t] = a2;
  }
}

__global__ void fill_g_kernel(const float* __restrict__ gout, f16* __restrict__ cat, int N) {
  size_t i = (size_t)blockIdx.x * blockDim.x + threadIdx.x;   // B*N*128
  int c = (int)(i & 127);
  size_t row = i >> 7;
  int b = (int)(row / N);
  cat[row * 256 + 128 + c] = (f16)gout[b * 128 + c];
}

__global__ void gpool_kernel(const float* __restrict__ feat2, const float* __restrict__ pw,
                             const float* __restrict__ pbp, float* __restrict__ y, int total) {
  int i = blockIdx.x * blockDim.x + threadIdx.x;
  if (i >= total) return;
  float nrm = 0.f, acc = pbp[0];
  for (int c = 0; c < 128; ++c) {
    float wv = pw[c];
    nrm += wv * wv;
    acc += feat2[(size_t)i * 128 + c] * wv;
  }
  y[i] = acc / sqrtf(nrm);
}

// exact-rank selection: top[b][rank] = index, rank by descending y (ties: lower idx)
__global__ __launch_bounds__(256)
void select_kernel(const float* __restrict__ y, int* __restrict__ top, int N, int kk) {
  __shared__ float ys[4096];
  int b = blockIdx.x;
  for (int i = threadIdx.x; i < N; i += 256) ys[i] = y[b * N + i];
  __syncthreads();
  for (int e = threadIdx.x; e < N; e += 256) {
    float ve = ys[e];
    int rank = 0;
    for (int j = 0; j < N; ++j) {
      float vj = ys[j];
      rank += (vj > ve) || (vj == ve && j < e);
    }
    if (rank < kk) top[b * kk + rank] = e;
  }
}

__global__ void gather_fd_kernel(const float* __restrict__ feat2, const float* __restrict__ y,
                                 const int* __restrict__ top, f16* __restrict__ fd,
                                 int N, int kk) {
  size_t i = (size_t)blockIdx.x * blockDim.x + threadIdx.x;  // B*kk*128
  int c = (int)(i & 127);
  size_t rr = i >> 7;
  int b = (int)(rr / kk);
  int e = top[rr];
  float s = 1.f / (1.f + expf(-y[b * N + e]));
  fd[rr * 128 + c] = (f16)(feat2[((size_t)b * N + e) * 128 + c] * s);
}

// wave-per-row down-MLP: 128->64->32->3, new_x = pos + r
__global__ __launch_bounds__(256)
void down_kernel(const float* __restrict__ x, const int* __restrict__ top,
                 const f16* __restrict__ fd,
                 const float* __restrict__ w1, const float* __restrict__ b1,
                 const float* __restrict__ w2, const float* __restrict__ b2,
                 const float* __restrict__ w3, const float* __restrict__ b3,
                 float* __restrict__ newx, int N, int kk) {
  int wid = threadIdx.x >> 5, lane = threadIdx.x & 31;
  size_t row = (size_t)blockIdx.x * 8 + wid;          // B*kk rows
  int b = (int)(row / kk);
  const f16* fr = fd + row * 128;
  float a0 = b1[lane], a1 = b1[lane + 32];
  for (int c = 0; c < 128; ++c) {
    float f = (float)fr[c];
    a0 += f * w1[c * 64 + lane];
    a1 += f * w1[c * 64 + lane + 32];
  }
  a0 = fmaxf(a0, 0.f); a1 = fmaxf(a1, 0.f);
  float a2 = b2[lane];
  for (int j = 0; j < 64; ++j) {
    float rj = (j < 32) ? __shfl(a0, j, 32) : __shfl(a1, j - 32, 32);
    a2 += rj * w2[j * 32 + lane];
  }
  a2 = fmaxf(a2, 0.f);
  float a3 = (lane < 3) ? b3[lane] : 0.f;
  for (int k = 0; k < 32; ++k) {
    float rk = __shfl(a2, k, 32);
    if (lane < 3) a3 += rk * w3[k * 3 + lane];
  }
  if (lane < 3) {
    int e = top[row];
    newx[row * 3 + lane] = x[((size_t)b * N + e) * 3 + lane] + a3;
  }
}

// xe rows (stride 160): cols 0..127 = fd, col 128 = mesh, cols 129..159 = 0
__global__ void build_xe_kernel(const f16* __restrict__ fd, f16* __restrict__ xe, int kk) {
  size_t i = (size_t)blockIdx.x * blockDim.x + threadIdx.x;   // (B*4*kk)*160
  int c = (int)(i % 160);
  size_t row = i / 160;
  size_t b = row / (4 * (size_t)kk);
  int rr = (int)(row % (4 * (size_t)kk));
  int t = rr / kk, r = rr % kk;
  f16 v = (f16)0;
  if (c < 128) v = fd[((size_t)b * kk + r) * 128 + c];
  else if (c == 128) v = (f16)(-0.2f + t * (0.4f / 3.0f));   // linspace(-0.2,0.2,4)
  xe[row * 160 + c] = v;
}

__global__ void final_kernel(const f16* __restrict__ h3, const float* __restrict__ w4,
                             const float* __restrict__ b4, const float* __restrict__ newx,
                             float* __restrict__ up, int kk) {
  size_t i = (size_t)blockIdx.x * blockDim.x + threadIdx.x;   // (B*4*kk)*3
  int c = (int)(i % 3);
  size_t row = i / 3;
  size_t b = row / (4 * (size_t)kk);
  int rr = (int)(row % (4 * (size_t)kk));
  int r = rr % kk;
  float acc = b4[c];
  const f16* hr = h3 + row * 64;
  for (int k = 0; k < 64; ++k) acc += (float)hr[k] * w4[k * 3 + c];
  up[row * 3 + c] = newx[((size_t)b * kk + r) * 3 + c] + acc;
}

// ------------------------------------------------------------------- launcher
extern "C" void kernel_launch(void* const* d_in, const int* in_sizes, int n_in,
                              void* d_out, int out_size, void* d_ws, size_t ws_size,
                              hipStream_t stream) {
  (void)in_sizes; (void)n_in; (void)out_size; (void)ws_size;
  const int B = 8, N = 4096, kk = 2048;
  const int BN = B * N;          // 32768 rows
  const int RF = B * 4 * kk;     // 65536 folding rows

  const float* x  = (const float*)d_in[0];
  const float* gf = (const float*)d_in[1];
  const float *cw[4], *cb[4], *cg[4], *cbe[4], *cm[4], *cv[4];
  for (int i = 0; i < 4; ++i) {
    cw[i]  = (const float*)d_in[2 + 6 * i + 0];
    cb[i]  = (const float*)d_in[2 + 6 * i + 1];
    cg[i]  = (const float*)d_in[2 + 6 * i + 2];
    cbe[i] = (const float*)d_in[2 + 6 * i + 3];
    cm[i]  = (const float*)d_in[2 + 6 * i + 4];
    cv[i]  = (const float*)d_in[2 + 6 * i + 5];
  }
  const float* wo = (const float*)d_in[26];  const float* bo = (const float*)d_in[27];
  const float* m1w1 = (const float*)d_in[28]; const float* m1b1 = (const float*)d_in[29];
  const float* m1w2 = (const float*)d_in[30]; const float* m1b2 = (const float*)d_in[31];
  const float* m2w1 = (const float*)d_in[32]; const float* m2b1 = (const float*)d_in[33];
  const float* m2w2 = (const float*)d_in[34]; const float* m2b2 = (const float*)d_in[35];
  const float* pw = (const float*)d_in[36];   const float* pb = (const float*)d_in[37];
  const float* dw1 = (const float*)d_in[38];  const float* db1 = (const float*)d_in[39];
  const float* dw2 = (const float*)d_in[40];  const float* db2 = (const float*)d_in[41];
  const float* dw3 = (const float*)d_in[42];  const float* db3 = (const float*)d_in[43];
  const float* fw1 = (const float*)d_in[44];  const float* fb1 = (const float*)d_in[45];
  const float* fw2 = (const float*)d_in[46];  const float* fb2 = (const float*)d_in[47];
  const float* fw3 = (const float*)d_in[48];  const float* fb3 = (const float*)d_in[49];
  const float* fw4 = (const float*)d_in[50];  const float* fb4 = (const float*)d_in[51];

  // ---- workspace layout (with region reuse) ----
  char* ws = (char*)d_ws;
  size_t off = 0;
  auto alloc = [&](size_t bytes) -> void* {
    off = (off + 255) & ~(size_t)255;
    void* p = ws + off;
    off += bytes;
    return p;
  };
  auto packed_elems = [](int K, int N) -> size_t {
    int Kp = ((K + 31) / 32) * 32;
    return (size_t)(N / 16) * (Kp / 32) * 512;
  };
  f16* wcp[4];
  const int cins[4] = {9, 41, 73, 105};
  for (int i = 0; i < 4; ++i) wcp[i] = (f16*)alloc(packed_elems(cins[i], 32) * 2);
  f16* wop    = (f16*)alloc(packed_elems(137, 128) * 2);
  f16* m2w1p  = (f16*)alloc(packed_elems(256, 256) * 2);
  f16* m2w2p  = (f16*)alloc(packed_elems(256, 128) * 2);
  f16* fw1p   = (f16*)alloc(packed_elems(129, 128) * 2);
  f16* fw2p   = (f16*)alloc(packed_elems(128, 128) * 2);
  f16* fw3p   = (f16*)alloc(packed_elems(128, 64) * 2);
  float* bnscale = (float*)alloc(4 * 32 * 4);
  float* bnshift = (float*)alloc(4 * 32 * 4);
  float* gout = (float*)alloc((size_t)B * 128 * 4);
  float* ybuf = (float*)alloc((size_t)BN * 4);
  int*   top  = (int*)alloc((size_t)B * kk * 4);
  int*   idxb = (int*)alloc((size_t)BN * 16 * 4);
  void* regionA = alloc((size_t)RF * 160 * 2);   // concat_in (BN*256) / xe (RF*160) / fh3
  void* regionB = alloc((size_t)RF * 128 * 2);   // mlp2 h1 / fold h1
  void* regionC = alloc((size_t)RF * 128 * 2);   // feat2 f32 (BN*128) / fold h2
  f16*  fd = (f16*)alloc((size_t)B * kk * 128 * 2);

  f16*   concat_in = (f16*)regionA;
  f16*   xe        = (f16*)regionA;
  f16*   fh3       = (f16*)regionA;
  f16*   m2h1      = (f16*)regionB;
  f16*   fh1       = (f16*)regionB;
  float* feat2     = (float*)regionC;
  f16*   fh2       = (f16*)regionC;

  float* out_newx = (float*)d_out;                  // (B,kk,3)
  float* out_up   = out_newx + (size_t)B * kk * 3;  // (B,4*kk,3)

  auto pack = [&](const float* W, f16* Wp, int K, int Nc) {
    int Kp = ((K + 31) / 32) * 32;
    size_t total = packed_elems(K, Nc);
    pack_b_kernel<<<(int)(total / 256), 256, 0, stream>>>(W, Wp, K, Nc, Kp);
  };

  // weight pre-pack (zero-padded, WMMA-native layout)
  for (int i = 0; i < 4; ++i) pack(cw[i], wcp[i], cins[i], 32);
  pack(wo, wop, 137, 128);
  pack(m2w1, m2w1p, 256, 256);
  pack(m2w2, m2w2p, 256, 128);
  pack(fw1, fw1p, 129, 128);
  pack(fw2, fw2p, 128, 128);
  pack(fw3, fw3p, 128, 64);
  for (int i = 0; i < 4; ++i)
    bn_fold_kernel<<<1, 32, 0, stream>>>(cb[i], cg[i], cbe[i], cm[i], cv[i],
                                         bnscale + i * 32, bnshift + i * 32);

  // 1) KNN
  knn_kernel<<<B * (N / 256), 256, 0, stream>>>(x, idxb, N);

  // 2) fused edge-conv encoder (WMMA) -> concat_in cols 0..127
  encoder_kernel<<<BN, 64, 0, stream>>>(x, idxb, wcp[0], wcp[1], wcp[2], wcp[3],
                                        bnscale, bnshift, wop, bo, concat_in, N);

  // 3) mlp1 + broadcast into concat_in cols 128..255
  mlp1_kernel<<<B, 256, 0, stream>>>(gf, m1w1, m1b1, m1w2, m1b2, gout);
  fill_g_kernel<<<(BN * 128) / 256, 256, 0, stream>>>(gout, concat_in, N);

  // 4) mlp2 (WMMA): 256->256 relu, 256->128 relu
  gemm16<<<((BN / 16) * (256 / 16)) / 4, 128, 0, stream>>>(
      concat_in, 256, m2w1p, m2b1, nullptr, m2h1, 256, BN, 256, 256, 1);
  gemm16<<<((BN / 16) * (128 / 16)) / 4, 128, 0, stream>>>(
      m2h1, 256, m2w2p, m2b2, feat2, nullptr, 128, BN, 256, 128, 1);

  // 5) GPool: projection, exact-rank top-half, gated gather
  gpool_kernel<<<BN / 256, 256, 0, stream>>>(feat2, pw, pb, ybuf, BN);
  select_kernel<<<B, 256, 0, stream>>>(ybuf, top, N, kk);
  gather_fd_kernel<<<(B * kk * 128) / 256, 256, 0, stream>>>(feat2, ybuf, top, fd, N, kk);

  // 6) down-MLP -> new_x (first output)
  down_kernel<<<(B * kk) / 8, 256, 0, stream>>>(x, top, fd, dw1, db1, dw2, db2,
                                                dw3, db3, out_newx, N, kk);

  // 7) folding (WMMA chain): xe(129, Kp=160) -> 128 -> 128 -> 64 -> 3
  build_xe_kernel<<<(int)(((size_t)RF * 160) / 256), 256, 0, stream>>>(fd, xe, kk);
  gemm16<<<((RF / 16) * (128 / 16)) / 4, 128, 0, stream>>>(
      xe, 160, fw1p, fb1, nullptr, fh1, 128, RF, 160, 128, 1);
  gemm16<<<((RF / 16) * (128 / 16)) / 4, 128, 0, stream>>>(
      fh1, 128, fw2p, fb2, nullptr, fh2, 128, RF, 128, 128, 1);
  gemm16<<<((RF / 16) * (64 / 16)) / 4, 128, 0, stream>>>(
      fh2, 128, fw3p, fb3, nullptr, fh3, 64, RF, 128, 64, 1);
  final_kernel<<<(int)(((size_t)RF * 3) / 256), 256, 0, stream>>>(fh3, fw4, fb4,
                                                                  out_newx, out_up, kk);
}
